// SubstructureAttentionDDI_71734543777918
// MI455X (gfx1250) — matrix operand order repaired and checked by reading, not compiled, not run on hardware
//
#include <hip/hip_runtime.h>
#include <hip/hip_bf16.h>

// Problem constants (match reference)
#define BB      256              // drug pairs
#define N_PER   40               // atoms per graph
#define HD      128              // hidden dim
#define N_ATOMS (BB * N_PER)     // 10240
#define NPAIR   (N_PER * N_PER)  // 1600
#define NGRP    (NPAIR / 16)     // 100 groups of 16 pairs
#define LDPH    136              // f16 LDS row stride: 64dw row + 4dw pad = 68dw

typedef __attribute__((ext_vector_type(16))) _Float16     v16h;
typedef __attribute__((ext_vector_type(8)))  _Float16     v8h;
typedef __attribute__((ext_vector_type(8)))  float        v8f;
typedef __attribute__((ext_vector_type(4)))  unsigned int v4u;
typedef __attribute__((ext_vector_type(8)))  int          v8i;
typedef __attribute__((ext_vector_type(4)))  int          v4i;

__device__ __forceinline__ float fast_tanh(float x) {
#if __has_builtin(__builtin_amdgcn_tanhf)
  return __builtin_amdgcn_tanhf(x);       // native v_tanh_f32
#else
  return tanhf(x);
#endif
}
__device__ __forceinline__ _Float16 fast_tanh_h(_Float16 x) {
#if __has_builtin(__builtin_amdgcn_tanh_f16)
  return __builtin_amdgcn_tanh_f16(x);    // native v_tanh_f16
#else
  return (_Float16)fast_tanh((float)x);
#endif
}

// ---------------------------------------------------------------------------
// TDM: 2D tile load global->LDS with per-row LDS padding.
// Descriptor per CDNA5 ISA ch.8:
//   group0: count=1 | lds_addr[63:32] | global_addr[120:64] | type=2[127:126]
//   group1: data_size code, pad_enable, pad_interval code, pad_amount code,
//           tensor_dim0=rowlen, tensor_dim1=rows, tile_dim0=rowlen,
//           tile_dim1=rows, tensor_dim0_stride=rowlen
// Issued by one wave; completion via s_wait_tensorcnt; EXEC ignored by TDM.
// ---------------------------------------------------------------------------
#if __has_builtin(__builtin_amdgcn_tensor_load_to_lds)
#define HAS_TDM 1
__device__ __forceinline__ void tdm_load_2d_padded(unsigned int lds_byte_off,
                                                   const void* gaddr,
                                                   unsigned int rows,
                                                   unsigned int rowlen_elems,
                                                   unsigned int dsz_code,
                                                   unsigned int pad_int_code,
                                                   unsigned int pad_amt_code) {
  const unsigned long long ga = (unsigned long long)gaddr;
  v4u g0;
  g0[0] = 1u;                                    // count=1 (valid user D#)
  g0[1] = lds_byte_off;                          // lds_addr
  g0[2] = (unsigned int)(ga & 0xFFFFFFFFu);      // global_addr[31:0]
  g0[3] = (unsigned int)((ga >> 32) & 0x1FFFFFFu) | (2u << 30);  // type=2
  v8i g1;
  g1[0] = (int)((dsz_code << 16) | (1u << 20) |
                (pad_int_code << 22) | (pad_amt_code << 25));
  g1[1] = (int)(rowlen_elems << 16);             // tensor_dim0[15:0] @ [63:48]
  g1[2] = (int)(rows << 16);                     // tensor_dim1[15:0] @ [95:80]
  g1[3] = (int)(rowlen_elems << 16);             // tile_dim0         @ [127:112]
  g1[4] = (int)rows;                             // tile_dim1         @ [143:128]
  g1[5] = (int)rowlen_elems;                     // tensor_dim0_stride[31:0]
  g1[6] = 0;
  g1[7] = 0;
  const v4i gz = {0, 0, 0, 0};
#if __clang_major__ >= 23
  const v8i gz8 = {0, 0, 0, 0, 0, 0, 0, 0};
  __builtin_amdgcn_tensor_load_to_lds(g0, g1, gz, gz, gz8, 0);
#else
  __builtin_amdgcn_tensor_load_to_lds(g0, g1, gz, gz, 0);
#endif
}
__device__ __forceinline__ void tdm_wait0() {
#if __has_builtin(__builtin_amdgcn_s_wait_tensorcnt)
  __builtin_amdgcn_s_wait_tensorcnt(0);
#else
  asm volatile("s_wait_tensorcnt 0x0" ::: "memory");
#endif
}
#else
#define HAS_TDM 0
#endif

// ---------------------------------------------------------------------------
// Kernel 0: pre-pack W1 (f32, row-major [2H,H]) into the wave32 WMMA B-operand
// layout as f16 (contiguous 32B per lane per K-step).
// B layout (32x16 f16): lane L holds col (L&15); b[t] = B[k0+(L>=16?16:0)+t][col].
// Wpk index: [z][ntile][kstep][lane][16]
// ---------------------------------------------------------------------------
__global__ __launch_bounds__(32) void pack_w1(const float* __restrict__ W1,
                                              _Float16* __restrict__ Wpk) {
  const int lane  = threadIdx.x;
  const int ntile = blockIdx.x;   // 0..7
  const int kstep = blockIdx.y;   // 0..3
  const int z     = blockIdx.z;   // 0..1
  const int ncol  = ntile * 16 + (lane & 15);
  const int k0    = kstep * 32 + (lane >> 4) * 16;
  const float* Wz = W1 + (size_t)z * HD * HD;
  _Float16* dst = Wpk + ((((size_t)z * 8 + ntile) * 4 + kstep) * 32 + lane) * 16;
#pragma unroll
  for (int t = 0; t < 16; ++t)
    dst[t] = (_Float16)Wz[(size_t)(k0 + t) * HD + ncol];
}

// ---------------------------------------------------------------------------
// Kernel 1: P1 = f16( h1 @ W1[:H] )        (z == 0)
//           P2 = f16( h2 @ W1[H:] + b1 )   (z == 1, bias folded in f32)
// One wave per 16x16 output tile; 4x v_wmma_f32_16x16x32_f16.
// ---------------------------------------------------------------------------
__global__ __launch_bounds__(32) void ddi_gemm_wmma(
    const float* __restrict__ h1, const float* __restrict__ h2,
    const _Float16* __restrict__ Wpk, const float* __restrict__ b1,
    _Float16* __restrict__ P1, _Float16* __restrict__ P2) {
  const int lane  = threadIdx.x;
  const int m0    = blockIdx.x * 16;
  const int ntile = blockIdx.y;
  const int n0    = ntile * 16;
  const int z     = blockIdx.z;

  const float* A = z ? h2 : h1;
  _Float16*    P = z ? P2 : P1;

  const int mrow = m0 + (lane & 15);
  const int ncol = n0 + (lane & 15);
  const int hi   = lane >> 4;        // lane half: 0 or 1
  const int aoff = hi * 16;          // A K sub-offset

  const float*    arow  = A + (size_t)mrow * HD;
  const _Float16* bbase = Wpk + ((((size_t)z * 8 + ntile) * 4) * 32 + lane) * 16;

  v8f acc = {};
#pragma unroll
  for (int k = 0; k < 4; ++k) {        // K-steps of 32
    const int k0 = k * 32;
    v16h a;
#pragma unroll
    for (int t = 0; t < 8; ++t) {
      a[t]     = (_Float16)arow[k0 + aoff + t];
      a[t + 8] = (_Float16)arow[k0 + 8 + aoff + t];
    }
    const v16h bm = *(const v16h*)(bbase + (size_t)k * 32 * 16);
    acc = __builtin_amdgcn_wmma_f32_16x16x32_f16(
        false, a, false, bm, (short)0, acc, false, false);
  }

  const float badd = z ? b1[ncol] : 0.0f;
#pragma unroll
  for (int v = 0; v < 8; ++v) {
    const int orow = m0 + v + hi * 8;  // D layout: VGPR v -> row v (+8 for hi)
    P[(size_t)orow * HD + ncol] = (_Float16)(acc[v] + badd);
  }
}

// ---------------------------------------------------------------------------
// Kernel 2: per drug-pair block.
//   logits[i,j] = sum_h tanh(p1[i,h] + p2'[j,h]) * W2[h] + b2
// The 128-term W2 dot is done on the matrix pipe: each wave owns 16 pairs;
// per K=32 step every lane builds its A-operand chunk as v_pk_add_f16 +
// v_tanh_f16 directly in WMMA A layout, and B is W2 in column 0 (else 0).
// Column 0 of D (lanes 0 and 16) holds the 16 logits.
// Then global softmax over 1600 scores; attn1 = row sums, attn2 = col sums.
// P tiles staged into LDS as f16 via TDM with hardware row padding
// (stride 68 dwords = 136 halves -> rows stay 16B-aligned, banks spread).
// ---------------------------------------------------------------------------
__global__ __launch_bounds__(256) void ddi_attn(
    const _Float16* __restrict__ P1, const _Float16* __restrict__ P2,
    const float* __restrict__ W2, const float* __restrict__ b2,
    float* __restrict__ out) {
  extern __shared__ __align__(16) unsigned char smem_raw[];
  _Float16* p1s = (_Float16*)smem_raw;                  // 40*136 halves (10880B)
  _Float16* p2s = p1s + N_PER * LDPH;                   // +10880B (16B aligned)
  float*    sc  = (float*)(p2s + N_PER * LDPH);         // 1600 floats
  float*    red = sc + NPAIR;                           // red[0..7], [8]=max, [9]=Z

  const int tid  = threadIdx.x;
  const int lane = tid & 31;
  const int wave = tid >> 5;
  const int b    = blockIdx.x;
  const int hi   = lane >> 4;            // lane half
  const int aoff = hi * 16;              // A K sub-offset (halves)
  const bool col0 = (lane & 15) == 0;    // lanes holding D column 0

  const _Float16* p1g = P1 + (size_t)b * N_PER * HD;
  const _Float16* p2g = P2 + (size_t)b * N_PER * HD;

#if HAS_TDM
  if (tid < 32) {                        // wave 0 issues both DMA descriptors
    // f16 (code 1), rows of 128 halves = 64 dw (pad_interval code 5),
    // pad 4 dwords (code 3) -> LDS row stride 68 dw = 136 halves.
    tdm_load_2d_padded(0u, p1g, N_PER, HD, 1u, 5u, 3u);
    tdm_load_2d_padded((unsigned int)(N_PER * LDPH * 2), p2g, N_PER, HD,
                       1u, 5u, 3u);
    tdm_wait0();
  }
#else
  for (int idx = tid; idx < N_PER * HD; idx += 256) {
    const int r = idx >> 7, c = idx & (HD - 1);
    p1s[r * LDPH + c] = p1g[idx];
    p2s[r * LDPH + c] = p2g[idx];
  }
#endif

  // B operand: W2 in column 0, zero elsewhere (per wave, kept in VGPRs).
  v16h bmk[4];
#pragma unroll
  for (int k = 0; k < 4; ++k)
#pragma unroll
    for (int t = 0; t < 16; ++t)
      bmk[k][t] = col0 ? (_Float16)W2[k * 32 + hi * 16 + t] : (_Float16)0.0f;

  __syncthreads();

  const float bias2 = b2[0];

  // ---- logits via tanh (trans pipe) + WMMA dot (matrix pipe) ----
  float lmax = -3.4e38f;
  for (int g = wave; g < NGRP; g += 8) {
    const int pg = g * 16 + (lane & 15);   // this lane's pair (A row)
    const int i  = pg / N_PER;
    const int j  = pg - i * N_PER;
    const _Float16* r1 = p1s + i * LDPH;
    const _Float16* r2 = p2s + j * LDPH;

    v8f acc = {};
#pragma unroll
    for (int k = 0; k < 4; ++k) {
      const int off0 = k * 32 + aoff;
      const v8h x0 = *(const v8h*)(r1 + off0)      + *(const v8h*)(r2 + off0);
      const v8h x1 = *(const v8h*)(r1 + off0 + 8) + *(const v8h*)(r2 + off0 + 8);
      v16h a;
#pragma unroll
      for (int t = 0; t < 8; ++t) {
        a[t]     = fast_tanh_h(x0[t]);
        a[t + 8] = fast_tanh_h(x1[t]);
      }
      acc = __builtin_amdgcn_wmma_f32_16x16x32_f16(
          false, a, false, bmk[k], (short)0, acc, false, false);
    }
    if (col0) {                            // lane0: rows 0-7, lane16: rows 8-15
#pragma unroll
      for (int v = 0; v < 8; ++v) {
        const float lg = acc[v] + bias2;
        sc[g * 16 + hi * 8 + v] = lg;
        lmax = fmaxf(lmax, lg);
      }
    }
  }

  // ---- block max (wave32 shfl + cross-wave LDS) ----
#pragma unroll
  for (int off = 16; off > 0; off >>= 1)
    lmax = fmaxf(lmax, __shfl_xor(lmax, off, 32));
  if (lane == 0) red[wave] = lmax;
  __syncthreads();
  if (tid == 0) {
    float m = red[0];
#pragma unroll
    for (int w = 1; w < 8; ++w) m = fmaxf(m, red[w]);
    red[8] = m;
  }
  __syncthreads();
  const float mx = red[8];

  // ---- exp + local sum ----
  float lsum = 0.0f;
  for (int p = tid; p < NPAIR; p += 256) {
    const float e = __expf(sc[p] - mx);
    sc[p] = e;
    lsum += e;
  }
#pragma unroll
  for (int off = 16; off > 0; off >>= 1)
    lsum += __shfl_xor(lsum, off, 32);
  __syncthreads();                         // red[0..7] reuse after max phase
  if (lane == 0) red[wave] = lsum;
  __syncthreads();
  if (tid == 0) {
    float s = 0.0f;
#pragma unroll
    for (int w = 0; w < 8; ++w) s += red[w];
    red[9] = s;
  }
  __syncthreads();
  const float inv = 1.0f / red[9];

  // ---- attn1 = row sums / Z ; attn2 = col sums / Z ----
  if (tid < N_PER) {
    float rs = 0.0f;
#pragma unroll 8
    for (int j = 0; j < N_PER; ++j) rs += sc[tid * N_PER + j];
    out[(size_t)b * N_PER + tid] = rs * inv;
  } else if (tid >= 64 && tid < 64 + N_PER) {
    const int j = tid - 64;
    float cs = 0.0f;
#pragma unroll 8
    for (int i = 0; i < N_PER; ++i) cs += sc[i * N_PER + j];
    out[(size_t)(BB * N_PER) + (size_t)b * N_PER + j] = cs * inv;
  }
}

// ---------------------------------------------------------------------------
// Launch
// Inputs (setup_inputs order): h1_atoms, h2_atoms, batch1, batch2, W1, b1, W2, b2
// batch1/batch2 unused (sorted equal segments -> implicit reshape).
// Workspace: P1 [10240*128 f16] ++ P2 [10240*128 f16] ++ Wpk [2*8*4*32*16 f16]
// Output: attn1 (10240 f32) ++ attn2 (10240 f32).
// ---------------------------------------------------------------------------
extern "C" void kernel_launch(void* const* d_in, const int* in_sizes, int n_in,
                              void* d_out, int out_size, void* d_ws, size_t ws_size,
                              hipStream_t stream) {
  const float* h1 = (const float*)d_in[0];
  const float* h2 = (const float*)d_in[1];
  const float* W1 = (const float*)d_in[4];
  const float* b1 = (const float*)d_in[5];
  const float* W2 = (const float*)d_in[6];
  const float* b2 = (const float*)d_in[7];

  _Float16* P1  = (_Float16*)d_ws;
  _Float16* P2  = P1 + (size_t)N_ATOMS * HD;
  _Float16* Wpk = P2 + (size_t)N_ATOMS * HD;

  pack_w1<<<dim3(8, 4, 2), dim3(32), 0, stream>>>(W1, Wpk);

  dim3 g1(N_ATOMS / 16, HD / 16, 2);   // (640, 8, 2) waves
  ddi_gemm_wmma<<<g1, dim3(32), 0, stream>>>(h1, h2, Wpk, b1, P1, P2);

  const size_t shmem = (size_t)(2 * N_PER * LDPH) * sizeof(_Float16) +
                       (size_t)(NPAIR + 10) * sizeof(float);
  ddi_attn<<<dim3(BB), dim3(256), shmem, stream>>>(P1, P2, W2, b2, (float*)d_out);
}
